// ParticlesNetwork_55448027791887
// MI455X (gfx1250) — matrix operand order repaired and checked by reading, not compiled, not run on hardware
//
#include <hip/hip_runtime.h>
#include <hip/hip_bf16.h>

// ---------------------------------------------------------------------------
// CDNA5 (gfx1250, wave32) implementation of the ParticlesNetwork reference.
// Both contractions of cts_conv are restructured into WMMA GEMMs
// (v_wmma_f32_16x16x32_f16). Dense branch / bias / concat / residual are
// folded into the second GEMM's K-dim and epilogue.
// ---------------------------------------------------------------------------

typedef __attribute__((ext_vector_type(16))) _Float16 v16h;
typedef __attribute__((ext_vector_type(8)))  _Float16 v8h;
typedef __attribute__((ext_vector_type(8)))  float    v8f;

static __device__ __forceinline__ float sgnf(float x) {
    return (x > 0.f) ? 1.f : ((x < 0.f) ? -1.f : 0.f);
}

// map_polar_sqr from the reference (per pair, scalar math)
static __device__ __forceinline__ void map_polar_sqr(float x, float y,
                                                     float& xo, float& yo) {
    const float c = 1.27323954473516268615f; // 4/pi
    float r  = sqrtf(x * x + y * y + 1e-9f);
    bool cond1 = (x == 0.f) && (y == 0.f);
    bool cond2 = (fabsf(y) <= fabsf(x)) && !cond1;
    float sx = (x == 0.f) ? 1.f : x;
    float sy = (y == 0.f) ? 1.f : y;
    float x2 = sgnf(x) * r;
    float y2 = c * sgnf(x) * r * atanf(y / sx);
    float x3 = c * sgnf(y) * r * atanf(x / sy);
    float y3 = sgnf(y) * r;
    xo = cond1 ? 0.f : (cond2 ? x2 : x3);
    yo = cond1 ? 0.f : (cond2 ? y2 : y3);
}

// axis_weights for size==4 (grid_sample bilinear, align_corners=False, zeros pad)
static __device__ __forceinline__ void axw4(float g, float w[4]) {
    float cc = 2.f * g + 1.5f;
    float i0 = floorf(cc);
    float f  = cc - i0;
    int k0 = (int)i0;
    w[0] = w[1] = w[2] = w[3] = 0.f;
    if (k0 >= 0 && k0 < 4)        w[k0]     = 1.f - f;
    if (k0 + 1 >= 0 && k0 + 1 < 4) w[k0 + 1] = f;
}

// ---------------------------------------------------------------------------
// Stage feats0 = [1, v, other_feats, 0-pad] (f16) into tail of T buffer.
// Layer-1 layout: T[n][64*80 + e], e in [0,96).
// ---------------------------------------------------------------------------
__global__ void feats0_to_T(const float* __restrict__ v,
                            const float* __restrict__ other,
                            _Float16* __restrict__ T, int KD2) {
    int idx = blockIdx.x * blockDim.x + threadIdx.x; // 512*96
    if (idx >= 512 * 96) return;
    int n = idx / 96, e = idx % 96;
    float val = 0.f;
    if (e == 0)       val = 1.f;
    else if (e < 4)   val = v[n * 3 + (e - 1)];
    else if (e < 66)  val = other[n * 62 + (e - 4)];
    T[(size_t)n * KD2 + 64 * 80 + e] = (_Float16)val;
}

// h = relu(prev) (f16) into tail of T buffer: T[n][64*64 + e], e in [0,64).
__global__ void relu_to_T(const float* __restrict__ x,
                          _Float16* __restrict__ T, int KD2) {
    int idx = blockIdx.x * blockDim.x + threadIdx.x; // 512*64
    if (idx >= 512 * 64) return;
    int n = idx / 64, e = idx % 64;
    T[(size_t)n * KD2 + 64 * 64 + e] = (_Float16)fmaxf(x[idx], 0.f);
}

// ---------------------------------------------------------------------------
// Build KfT[o][KD2] (f16, o-major so B fragments are contiguous in K) plus bias.
// Row r < 64*I_PAD: conv part, r = g*I_PAD + i  -> kern[o][i][g]
// Row r >= 64*I_PAD: dense part -> W[o][i]
// split==1 (layer 1): o<32 conv-only (k_fluid), o>=32 dense-only (Wf), bias
// only on the dense half (matches concat([oc, od])).
// ---------------------------------------------------------------------------
__global__ void build_kft(const float* __restrict__ kern,
                          const float* __restrict__ W,
                          const float* __restrict__ bvec,
                          _Float16* __restrict__ KfT,
                          float* __restrict__ bias,
                          int O, int I, int I_PAD, int KD2, int split) {
    int total = O * KD2;
    int convBase = 64 * I_PAD;
    for (int idx = blockIdx.x * blockDim.x + threadIdx.x; idx < total;
         idx += gridDim.x * blockDim.x) {
        int o = idx / KD2, r = idx % KD2;
        float val = 0.f;
        if (r < convBase) {
            int g = r / I_PAD, i = r % I_PAD;
            bool convEn = (!split) || (o < 32);
            if (i < I && convEn)
                val = kern[(size_t)o * I * 64 + (size_t)i * 64 + g];
        } else {
            int i = r - convBase;
            if (split) {
                if (o >= 32 && i < I) val = W[(o - 32) * I + i];
            } else {
                if (i < I) val = W[o * I + i];
            }
        }
        KfT[idx] = (_Float16)val;
    }
    if (blockIdx.x == 0 && (int)threadIdx.x < O) {
        int o = threadIdx.x;
        bias[o] = split ? ((o >= 32) ? bvec[o - 32] : 0.f) : bvec[o];
    }
}

// ---------------------------------------------------------------------------
// conv_gather: one block per query particle m (512 blocks, 4 wave32s).
// t[g=64, i=I_PAD] = sum_n w[n,g] * (h[n,i]*att[n]); K-loop over n in 32-chunks.
// Wave wv owns g-rows [16wv,16wv+16); A = w^T tile (16x32 f16 from LDS),
// B = fa tile (32x16 f16 from LDS); I_TILES accumulators of v8f.
// Writes t (f16) into head of T: T[m][g*I_PAD + i].
// ---------------------------------------------------------------------------
template <int I_TILES>
__global__ __launch_bounds__(128) void conv_gather(
    const float* __restrict__ p, const float* __restrict__ mask,
    _Float16* __restrict__ T, int KD2) {
    constexpr int I_PAD = I_TILES * 16;
    const int m    = blockIdx.x;
    const int tid  = threadIdx.x;
    const int lane = tid & 31;
    const int wv   = tid >> 5;

    __shared__ __align__(32) _Float16 wT[64 * 32];       // [g][n]
    __shared__ __align__(32) _Float16 faT[I_PAD * 32];   // [i][n]
    __shared__ float aw[32 * 12];                        // wz[4],wy[4],wx[4] per n
    __shared__ float attS[32];

    const float pmx = p[m * 3 + 0], pmy = p[m * 3 + 1], pmz = p[m * 3 + 2];
    const int hOff = 64 * I_PAD;

    v8f acc[I_TILES];
#pragma unroll
    for (int t = 0; t < I_TILES; ++t)
#pragma unroll
        for (int e = 0; e < 8; ++e) acc[t][e] = 0.f;

    for (int n0 = 0; n0 < 512; n0 += 32) {
        // stage 1: pairwise geometry (wave 0)
        if (tid < 32) {
            int n = n0 + tid;
            float rx = (p[n * 3 + 0] - pmx) * (1.f / 40.f);
            float ry = (p[n * 3 + 1] - pmy) * (1.f / 40.f);
            float rz = (p[n * 3 + 2] - pmz) * (1.f / 40.f);
            float r2 = rx * rx + ry * ry + rz * rz;
            float a1 = 1.f - r2;
            float att = fmaxf(a1 * a1 * a1, 0.f) * mask[n];
            float xs, ys;
            map_polar_sqr(rx, ry, xs, ys);
            float wz[4], wy[4], wx[4];
            axw4(rz, wz);  // z -> kD
            axw4(ys, wy);  // y -> kH
            axw4(xs, wx);  // x -> kW
#pragma unroll
            for (int k = 0; k < 4; ++k) {
                aw[tid * 12 + k]     = wz[k];
                aw[tid * 12 + 4 + k] = wy[k];
                aw[tid * 12 + 8 + k] = wx[k];
            }
            attS[tid] = att;
        }
        __syncthreads();

        // stage 2: dense trilinear weights wT[g][n], g = z*16 + y*4 + x
#pragma unroll
        for (int e = 0; e < 16; ++e) {
            int idx = tid + e * 128;              // 0..2047
            int g = idx >> 5, n = idx & 31;
            int gz = g >> 4, gy = (g >> 2) & 3, gx = g & 3;
            wT[g * 32 + n] =
                (_Float16)(aw[n * 12 + gz] * aw[n * 12 + 4 + gy] * aw[n * 12 + 8 + gx]);
        }
        // faT[i][n] = h[n][i] * att[n]   (h read from T tail, f16)
        for (int idx = tid; idx < I_PAD * 32; idx += 128) {
            int i = idx >> 5, n = idx & 31;
            float hv = (float)T[(size_t)(n0 + n) * KD2 + hOff + i];
            faT[i * 32 + n] = (_Float16)(hv * attS[n]);
        }
        __syncthreads();

        // stage 3: WMMA accumulate. A: 16x32 f16 (lanes 0-15 K{0-7,16-23},
        // lanes 16-31 K{8-15,24-31}); B: lane = column, same K split as A.
        {
            int gRow = wv * 16 + (lane & 15);
            int kb   = (lane < 16) ? 0 : 8;
            v8h alo = *(const v8h*)&wT[gRow * 32 + kb];
            v8h ahi = *(const v8h*)&wT[gRow * 32 + kb + 16];
            v16h a;
#pragma unroll
            for (int e = 0; e < 8; ++e) { a[e] = alo[e]; a[8 + e] = ahi[e]; }
            int kOff = kb;
#pragma unroll
            for (int t = 0; t < I_TILES; ++t) {
                int iCol = t * 16 + (lane & 15);
                v8h blo = *(const v8h*)&faT[iCol * 32 + kOff];
                v8h bhi = *(const v8h*)&faT[iCol * 32 + kOff + 16];
                v16h b;
#pragma unroll
                for (int e = 0; e < 8; ++e) { b[e] = blo[e]; b[8 + e] = bhi[e]; }
                acc[t] = __builtin_amdgcn_wmma_f32_16x16x32_f16(
                    false, a, false, b, (short)0, acc[t], false, false);
            }
        }
        __syncthreads();
    }

    // write t tiles to T head (C layout: VGPR v -> M=v / v+8; N = lane&15)
    int rowAdd = (lane < 16) ? 0 : 8;
    int colN   = lane & 15;
#pragma unroll
    for (int t = 0; t < I_TILES; ++t)
#pragma unroll
        for (int v = 0; v < 8; ++v) {
            int g = wv * 16 + v + rowAdd;
            int i = t * 16 + colN;
            T[(size_t)m * KD2 + g * I_PAD + i] = (_Float16)acc[t][v];
        }
}

// ---------------------------------------------------------------------------
// gemm_out: out[512, O] = T[512, KD2] x KfT^T + bias (+ residual).
// 32 blocks of 4 wave32s; block = 16-row m-tile, wave = 16-col o-tile.
// A read straight from T (row-major f16), B from KfT (o-major f16); both
// 128-bit-aligned fragment loads. Fused epilogue.
// ---------------------------------------------------------------------------
__global__ __launch_bounds__(128) void gemm_out(
    const _Float16* __restrict__ T, int KD2,
    const _Float16* __restrict__ KfT,
    const float* __restrict__ bias,
    const float* __restrict__ resid,  // nullptr -> no residual
    float* __restrict__ out, int O) {
    const int mtile = blockIdx.x;          // 0..31
    const int tid   = threadIdx.x;
    const int lane  = tid & 31;
    const int otile = tid >> 5;
    const int O_T   = (O + 15) >> 4;
    if (otile >= O_T) return;  // wave-uniform exit; no barriers used here

    v8f acc;
#pragma unroll
    for (int e = 0; e < 8; ++e) acc[e] = 0.f;

    const _Float16* Trow =
        T + (size_t)(mtile * 16 + (lane & 15)) * KD2 + ((lane < 16) ? 0 : 8);
    const _Float16* Brow =
        KfT + (size_t)(otile * 16 + (lane & 15)) * KD2 + ((lane < 16) ? 0 : 8);

    for (int k = 0; k < KD2; k += 32) {
        __builtin_prefetch(Brow + k + 512, 0, 0);  // global_prefetch_b8
        v8h alo = *(const v8h*)(Trow + k);
        v8h ahi = *(const v8h*)(Trow + k + 16);
        v16h a;
#pragma unroll
        for (int e = 0; e < 8; ++e) { a[e] = alo[e]; a[8 + e] = ahi[e]; }
        v8h blo = *(const v8h*)(Brow + k);
        v8h bhi = *(const v8h*)(Brow + k + 16);
        v16h b;
#pragma unroll
        for (int e = 0; e < 8; ++e) { b[e] = blo[e]; b[8 + e] = bhi[e]; }
        acc = __builtin_amdgcn_wmma_f32_16x16x32_f16(
            false, a, false, b, (short)0, acc, false, false);
    }

    int rowAdd = (lane < 16) ? 0 : 8;
    int col    = otile * 16 + (lane & 15);
    if (col < O) {
#pragma unroll
        for (int v = 0; v < 8; ++v) {
            int row   = mtile * 16 + v + rowAdd;
            float val = acc[v] + bias[col];
            if (resid) val += resid[row * O + col];
            out[row * O + col] = val;
        }
    }
}

// ---------------------------------------------------------------------------
extern "C" void kernel_launch(void* const* d_in, const int* in_sizes, int n_in,
                              void* d_out, int out_size, void* d_ws, size_t ws_size,
                              hipStream_t stream) {
    (void)in_sizes; (void)n_in; (void)out_size; (void)ws_size;
    const float* p     = (const float*)d_in[0];
    const float* v     = (const float*)d_in[1];
    const float* other = (const float*)d_in[2];
    const float* mask  = (const float*)d_in[3];
    const float* kf = (const float*)d_in[4];
    const float* Wf = (const float*)d_in[5];
    const float* bf = (const float*)d_in[6];
    const float* kL[3] = {(const float*)d_in[7], (const float*)d_in[10], (const float*)d_in[13]};
    const float* WL[3] = {(const float*)d_in[8], (const float*)d_in[11], (const float*)d_in[14]};
    const float* bL[3] = {(const float*)d_in[9], (const float*)d_in[12], (const float*)d_in[15]};
    const float* k4 = (const float*)d_in[16];
    const float* W4 = (const float*)d_in[17];
    const float* b4 = (const float*)d_in[18];

    // workspace carve-out (~6.3 MB)
    size_t off = 0;
    auto carve = [&](size_t bytes) {
        void* r = (char*)d_ws + off;
        off += (bytes + 255) & ~(size_t)255;
        return r;
    };
    const int KD2_1 = 64 * 80 + 96;   // 5216 (layer 1: I_PAD=80, dense pad 96)
    const int KD2_N = 64 * 64 + 64;   // 4160 (layers 2-5)
    _Float16* T    = (_Float16*)carve((size_t)512 * KD2_1 * 2);
    _Float16* KfT  = (_Float16*)carve((size_t)64 * KD2_1 * 2);
    float*    bias = (float*)carve(64 * 4);
    float*    bufA = (float*)carve((size_t)512 * 64 * 4);
    float*    bufB = (float*)carve((size_t)512 * 64 * 4);

    // ---- Layer 1 (k_fluid conv o<32, Wf dense o>=32, concat) ----
    feats0_to_T<<<(512 * 96 + 255) / 256, 256, 0, stream>>>(v, other, T, KD2_1);
    build_kft<<<512, 256, 0, stream>>>(kf, Wf, bf, KfT, bias, 64, 66, 80, KD2_1, 1);
    conv_gather<5><<<512, 128, 0, stream>>>(p, mask, T, KD2_1);
    gemm_out<<<32, 128, 0, stream>>>(T, KD2_1, KfT, bias, nullptr, bufA, 64);

    // ---- Layers 2-4 (residual) ----
    float* prev = bufA;
    float* next = bufB;
    for (int l = 0; l < 3; ++l) {
        relu_to_T<<<(512 * 64 + 255) / 256, 256, 0, stream>>>(prev, T, KD2_N);
        build_kft<<<512, 256, 0, stream>>>(kL[l], WL[l], bL[l], KfT, bias, 64, 64, 64, KD2_N, 0);
        conv_gather<4><<<512, 128, 0, stream>>>(p, mask, T, KD2_N);
        gemm_out<<<32, 128, 0, stream>>>(T, KD2_N, KfT, bias, prev, next, 64);
        float* tmp = prev; prev = next; next = tmp;
    }

    // ---- Layer 5 (O=6, no residual) -> d_out ----
    relu_to_T<<<(512 * 64 + 255) / 256, 256, 0, stream>>>(prev, T, KD2_N);
    build_kft<<<512, 256, 0, stream>>>(k4, W4, b4, KfT, bias, 6, 64, 64, KD2_N, 0);
    conv_gather<4><<<512, 128, 0, stream>>>(p, mask, T, KD2_N);
    gemm_out<<<32, 128, 0, stream>>>(T, KD2_N, KfT, bias, nullptr, (float*)d_out, 6);
}
